// CausalSelfAttention_61306363183634
// MI455X (gfx1250) — matrix operand order repaired and checked by reading, not compiled, not run on hardware
//
#include <hip/hip_runtime.h>

// ---------------------------------------------------------------------------
// CDNA5 (gfx1250) causal self-attention forward.
// Matrix math: v_wmma_f32_16x16x32_bf16 (wave32). K/V staging: TDM
// (tensor_load_to_lds + TENSORcnt), double-buffered LDS.
// ---------------------------------------------------------------------------

typedef __bf16 bf16;
typedef __attribute__((ext_vector_type(16))) __bf16 v16bf;
typedef __attribute__((ext_vector_type(8)))  __bf16 v8bf;
typedef __attribute__((ext_vector_type(4)))  __bf16 v4bf;
typedef __attribute__((ext_vector_type(8)))  float  v8f;
typedef __attribute__((ext_vector_type(4)))  float  v4f;
typedef __attribute__((ext_vector_type(4)))  unsigned int v4u;
typedef __attribute__((ext_vector_type(8)))  int    v8i;
typedef __attribute__((ext_vector_type(4)))  int    v4i;

#if defined(__has_builtin)
#  if __has_builtin(__builtin_amdgcn_tensor_load_to_lds)
#    define USE_TDM 1
#  endif
#endif
#ifndef USE_TDM
#  define USE_TDM 0
#endif

// Build a 16-element bf16 fragment from two contiguous 16-byte chunks.
__device__ __forceinline__ v16bf pack16(const bf16* lo, const bf16* hi) {
    v8bf l = *(const v8bf*)lo;
    v8bf h = *(const v8bf*)hi;
    v16bf r;
#pragma unroll
    for (int i = 0; i < 8; ++i) { r[i] = l[i]; r[i + 8] = h[i]; }
    return r;
}

__device__ __forceinline__ v8f wmma_bf16(v16bf a, v16bf b, v8f c) {
    return __builtin_amdgcn_wmma_f32_16x16x32_bf16(false, a, false, b,
                                                   (short)0, c, false, false);
}

// Generic LDS pointer -> LDS byte offset (aperture maps addr[31:0] to LDS).
__device__ __forceinline__ unsigned int lds_off(const void* p) {
    return (unsigned int)(uintptr_t)p;
}

#if USE_TDM
// Issue one TDM 2D tile load (global -> LDS) per ISA D# layout (ch.8).
// data_size = 2 bytes. Pads pad_amt_code+1 DWORDs into LDS every
// 2^(pad_int_code+1) DWORDs (reproduces our padded LDS row strides).
__device__ __forceinline__ void tdm_load_2d(
    const void* gaddr, unsigned int ldsaddr,
    unsigned int tensor_d0, unsigned int tensor_d1, unsigned long long stride0,
    unsigned int tile_d0, unsigned int tile_d1,
    unsigned int pad_int_code, unsigned int pad_amt_code) {
    unsigned long long ga = (unsigned long long)(uintptr_t)gaddr;
    v4u g0;
    g0[0] = 1u;                                         // count=1, user D#
    g0[1] = ldsaddr;                                    // lds_addr
    g0[2] = (unsigned int)ga;                           // global_addr[31:0]
    g0[3] = (unsigned int)((ga >> 32) & 0x1FFFFFFu) | (2u << 30);  // type=2
    v8i g1;
    g1[0] = (int)((1u << 16) |                          // data_size = 2B
                  (1u << 20) |                          // pad_enable
                  (pad_int_code << 22) | (pad_amt_code << 25));
    g1[1] = (int)((tensor_d0 & 0xFFFFu) << 16);         // tensor_dim0[15:0]
    g1[2] = (int)(((tensor_d0 >> 16) & 0xFFFFu) | ((tensor_d1 & 0xFFFFu) << 16));
    g1[3] = (int)(((tensor_d1 >> 16) & 0xFFFFu) | ((tile_d0 & 0xFFFFu) << 16));
    g1[4] = (int)(tile_d1 & 0xFFFFu);                   // tile_dim1; tile_dim2=0
    g1[5] = (int)(unsigned int)(stride0 & 0xFFFFFFFFu); // dim0_stride[31:0]
    g1[6] = (int)(unsigned int)((stride0 >> 32) & 0xFFFFu);
    g1[7] = 0;
    v4i gz = {0, 0, 0, 0};
#if __clang_major__ >= 23
    v8i gz8 = {0, 0, 0, 0, 0, 0, 0, 0};
    __builtin_amdgcn_tensor_load_to_lds(g0, g1, gz, gz, gz8, 0);
#else
    __builtin_amdgcn_tensor_load_to_lds(g0, g1, gz, gz, 0);
#endif
}
#endif

// ---------------------------------------------------------------------------
// fp32 -> bf16 weight conversion (one-time per launch)
// ---------------------------------------------------------------------------
__global__ void cvt_f32_to_bf16(const float* __restrict__ src,
                                bf16* __restrict__ dst, int n) {
    int i = (blockIdx.x * blockDim.x + threadIdx.x) * 4;
    if (i + 3 < n) {
        v4f x = *(const v4f*)(src + i);
        v4bf y;
#pragma unroll
        for (int e = 0; e < 4; ++e) y[e] = (bf16)x[e];
        *(v4bf*)(dst + i) = y;
    }
}

// ---------------------------------------------------------------------------
// Tiled GEMM: out[M,N] = X(fp32)[M,K] * W(bf16)[K,N] + bias.
// Block tile 128x128, K-step 32, double-buffered LDS; 8 waves x (4x2 frags).
// EPI==0: fp32 row-major.  EPI==1: QKV scatter (Q,K [B,H,C,hd]; V^T [B,H,hd,C]).
// ---------------------------------------------------------------------------
constexpr int BM = 128, BN = 128, BK = 32;
constexpr int ASTR = 40;
constexpr int BSTR = 40;

template <int EPI>
__global__ __launch_bounds__(256) void gemm_bf16(
    const float* __restrict__ X, const bf16* __restrict__ W,
    const float* __restrict__ bias, float* __restrict__ outF,
    bf16* __restrict__ Qb, bf16* __restrict__ Kb, bf16* __restrict__ VTb,
    int M, int N, int Kdim) {
    __shared__ bf16 As[2][BM * ASTR];   // [m][k]
    __shared__ bf16 Bs[2][BN * BSTR];   // [n][k] (W transposed)

    const int tid  = threadIdx.x;
    const int w    = tid >> 5, lane = tid & 31;
    const int ln   = lane & 15, lh = lane >> 4;
    const int wm   = w >> 2, wn = w & 3;
    const int m0   = blockIdx.y * BM, n0 = blockIdx.x * BN;

    v8f zero = {};
    v8f acc[4][2];
#pragma unroll
    for (int i = 0; i < 4; ++i)
#pragma unroll
        for (int j = 0; j < 2; ++j) acc[i][j] = zero;

    const int arow = tid >> 3;
    const int acol = (tid & 7) * 4;
    const int nkb  = Kdim / BK;

    v4f  aReg[4];
    v8bf bReg[2];

    auto load_global = [&](int kb) {
#pragma unroll
        for (int rr = 0; rr < 4; ++rr)
            aReg[rr] = *(const v4f*)(X + (size_t)(m0 + arow + 32 * rr) * Kdim +
                                     kb * BK + acol);
#pragma unroll
        for (int it = 0; it < 2; ++it) {
            int cid = tid + 256 * it;
            bReg[it] = *(const v8bf*)(W + (size_t)(kb * BK + (cid >> 4)) * N +
                                      n0 + (cid & 15) * 8);
        }
    };
    auto store_lds = [&](int buf) {
#pragma unroll
        for (int rr = 0; rr < 4; ++rr) {
            v4bf xb;
#pragma unroll
            for (int e = 0; e < 4; ++e) xb[e] = (bf16)aReg[rr][e];
            *(v4bf*)(As[buf] + (arow + 32 * rr) * ASTR + acol) = xb;
        }
#pragma unroll
        for (int it = 0; it < 2; ++it) {
            int cid = tid + 256 * it;
            int kk = cid >> 4, n8 = (cid & 15) * 8;
#pragma unroll
            for (int e = 0; e < 8; ++e) Bs[buf][(n8 + e) * BSTR + kk] = bReg[it][e];
        }
    };

    load_global(0);
    store_lds(0);
    __syncthreads();

    for (int kb = 0; kb < nkb; ++kb) {
        if (kb + 1 < nkb) load_global(kb + 1);   // flies during WMMAs

        const bf16* Ab = As[kb & 1];
        const bf16* Bb = Bs[kb & 1];
        v16bf a[4], bf[2];
#pragma unroll
        for (int i = 0; i < 4; ++i) {
            const bf16* p = Ab + (wm * 64 + i * 16 + ln) * ASTR;
            a[i] = pack16(p + 8 * lh, p + 16 + 8 * lh);
        }
#pragma unroll
        for (int j = 0; j < 2; ++j) {
            const bf16* p = Bb + (wn * 32 + j * 16 + ln) * BSTR + 8 * lh;
            bf[j] = pack16(p, p + 16);
        }
#pragma unroll
        for (int i = 0; i < 4; ++i)
#pragma unroll
            for (int j = 0; j < 2; ++j)
                acc[i][j] = wmma_bf16(a[i], bf[j], acc[i][j]);

        if (kb + 1 < nkb) {
            __syncthreads();
            store_lds((kb + 1) & 1);
            __syncthreads();
        }
    }

#pragma unroll
    for (int i = 0; i < 4; ++i) {
#pragma unroll
        for (int j = 0; j < 2; ++j) {
            int gn = n0 + wn * 32 + j * 16 + ln;
            float bv = bias[gn];
#pragma unroll
            for (int r = 0; r < 8; ++r) {
                int gm  = m0 + wm * 64 + i * 16 + r + 8 * lh;
                float v = acc[i][j][r] + bv;
                if constexpr (EPI == 0) {
                    outF[(size_t)gm * N + gn] = v;
                } else {
                    int bb = gm >> 11, c = gm & 2047;
                    int s = gn >> 10, idx = gn & 1023;
                    int hh = idx >> 6, dd = idx & 63;
                    size_t bh = (size_t)(bb * 16 + hh);
                    if (s == 0)      Qb[(bh * 2048 + c) * 64 + dd] = (bf16)v;
                    else if (s == 1) Kb[(bh * 2048 + c) * 64 + dd] = (bf16)v;
                    else             VTb[(bh * 64 + dd) * 2048 + c] = (bf16)v;
                }
            }
        }
    }
}

// ---------------------------------------------------------------------------
// Flash attention forward. Grid: (C/128, B*H). 256 threads = 8 waves; wave w
// owns q rows [qbase, qbase+16). K/V^T blocks DMA'd into double-buffered LDS
// by the Tensor Data Mover (wave 0 issues descriptors, TENSORcnt tracks,
// barrier releases consumers); block kb+1 streams while kb computes.
// With 128-row q-tiles and 128-col k-blocks every wave needs exactly
// blockIdx.x+1 k-blocks, so there is no per-wave causal divergence.
// ---------------------------------------------------------------------------
__global__ __launch_bounds__(256) void attn_fwd(
    const bf16* __restrict__ Q, const bf16* __restrict__ K,
    const bf16* __restrict__ VT, const float* __restrict__ mask,
    float* __restrict__ attnO) {
    constexpr int C = 2048, HD = 64;
    constexpr int KST = 72;    // 64 + 8 pad (TDM: pad 4 dw every 32 dw)
    constexpr int VST = 136;   // 128 + 8 pad (TDM: pad 4 dw every 64 dw)
    constexpr int PST = 136;
    __shared__ bf16 Ks[2][128 * KST];    // 36 KB : K blocks   [k][d]
    __shared__ bf16 Vs[2][HD * VST];     // 34 KB : V^T blocks [d][k]
    __shared__ bf16 Plds[8][16 * PST];   // 34 KB : per-wave P staging

    const int tid = threadIdx.x;
    const int w = tid >> 5, lane = tid & 31;
    const int ln = lane & 15, lh = lane >> 4;
    const int qbase = blockIdx.x * 128 + w * 16;
    const int b = blockIdx.y >> 4, h = blockIdx.y & 15;
    const size_t bh = blockIdx.y;

    const bf16* Kg = K + bh * (size_t)C * HD;
    const bf16* Vg = VT + bh * (size_t)HD * C;

#if USE_TDM
    auto stage = [&](int kb, int buf) {
        const int kbase = kb * 128;
        // K block: 128 rows x 64 cols of [C,64] tensor (rows contiguous)
        tdm_load_2d(Kg + (size_t)kbase * HD, lds_off(&Ks[buf][0]),
                    /*tensor_d0=*/64, /*tensor_d1=*/C, /*stride0=*/64,
                    /*tile_d0=*/64, /*tile_d1=*/128,
                    /*pad_int(32dw)=*/4, /*pad_amt(4dw)=*/3);
        // V^T block: 64 rows x 128 cols of [64,C] tensor
        tdm_load_2d(Vg + kbase, lds_off(&Vs[buf][0]),
                    /*tensor_d0=*/C, /*tensor_d1=*/HD, /*stride0=*/C,
                    /*tile_d0=*/128, /*tile_d1=*/64,
                    /*pad_int(64dw)=*/5, /*pad_amt(4dw)=*/3);
    };
#else
    auto stage = [&](int kb, int buf) {
        const int kbase = kb * 128;
#pragma unroll
        for (int it = 0; it < 4; ++it) {
            int cid = tid + 256 * it;
            int kk = cid >> 3, kc = (cid & 7) * 8;
            *(v8bf*)(&Ks[buf][0] + kk * KST + kc) =
                *(const v8bf*)(Kg + (size_t)(kbase + kk) * HD + kc);
            int d = cid >> 4, vc = (cid & 15) * 8;
            *(v8bf*)(&Vs[buf][0] + d * VST + vc) =
                *(const v8bf*)(Vg + (size_t)d * C + kbase + vc);
        }
    };
#endif

    // Q fragments for both K-steps over hd=64 (A layout: row=ln, d contiguous)
    v16bf aq[2];
    const bf16* qp = Q + (bh * C + qbase + ln) * HD;
#pragma unroll
    for (int ks = 0; ks < 2; ++ks)
        aq[ks] = pack16(qp + ks * 32 + 8 * lh, qp + ks * 32 + 16 + 8 * lh);

    float mrun[8], lrun[8];
    v8f zero = {};
    v8f o[4];
#pragma unroll
    for (int r = 0; r < 8; ++r) { mrun[r] = -__builtin_inff(); lrun[r] = 0.f; }
#pragma unroll
    for (int t = 0; t < 4; ++t) o[t] = zero;

    const float scaling = 0.125f;       // hd^-0.5
    const int nkb = blockIdx.x + 1;     // causal block bound (same for all waves)
    bf16* pw = &Plds[w][0];

#if USE_TDM
    if (w == 0) stage(0, 0);
#else
    stage(0, 0);
#endif

    for (int kb = 0; kb < nkb; ++kb) {
        const int buf = kb & 1;
#if USE_TDM
        if (w == 0) {
            if (kb + 1 < nkb) {
                stage(kb + 1, buf ^ 1);                 // DMA next block
                __builtin_amdgcn_s_wait_tensorcnt(2);   // kb's pair landed
            } else {
                __builtin_amdgcn_s_wait_tensorcnt(0);
            }
        }
        __syncthreads();    // release consumers of buf
#else
        __syncthreads();
        if (kb + 1 < nkb) stage(kb + 1, buf ^ 1);
#endif
        const bf16* Kbuf = &Ks[buf][0];
        const bf16* Vbuf = &Vs[buf][0];
        const int kbase = kb * 128;

        // prefetch next mask block rows toward L2 while we compute
        if (kb + 1 < nkb)
            __builtin_prefetch(
                mask + ((size_t)b * C + qbase + (lane >> 1)) * C + kbase + 128 +
                    (lane & 1) * 64, 0, 0);

        // ---- S = Q K^T (16 x 128 per wave), pipelined fragment loads ----
        v8f s[8];
#pragma unroll
        for (int j = 0; j < 8; ++j) s[j] = zero;

        v16bf bk0 = pack16(Kbuf + ln * KST + 8 * lh,
                           Kbuf + ln * KST + 8 * lh + 16);
        v16bf bk1 = pack16(Kbuf + ln * KST + 32 + 8 * lh,
                           Kbuf + ln * KST + 32 + 8 * lh + 16);
#pragma unroll
        for (int j = 0; j < 8; ++j) {
            v16bf nk0, nk1;
            if (j < 7) {
                const bf16* p = Kbuf + ((j + 1) * 16 + ln) * KST + 8 * lh;
                nk0 = pack16(p, p + 16);
                nk1 = pack16(p + 32, p + 48);
            }
            s[j] = wmma_bf16(aq[0], bk0, s[j]);
            s[j] = wmma_bf16(aq[1], bk1, s[j]);
            bk0 = nk0;
            bk1 = nk1;
        }

        // ---- scale + attention mask + causal; per-row max ----
        float tmax[8];
#pragma unroll
        for (int r = 0; r < 8; ++r) tmax[r] = -__builtin_inff();
#pragma unroll
        for (int j = 0; j < 8; ++j) {
            const int k = kbase + j * 16 + ln;
#pragma unroll
            for (int r = 0; r < 8; ++r) {
                const int q = qbase + r + 8 * lh;
                float v;
                if (k <= q)
                    v = s[j][r] * scaling + mask[((size_t)b * C + q) * C + k];
                else
                    v = -__builtin_inff();
                s[j][r] = v;
                tmax[r] = fmaxf(tmax[r], v);
            }
        }
#pragma unroll
        for (int r = 0; r < 8; ++r) {
#pragma unroll
            for (int msk = 1; msk <= 8; msk <<= 1)
                tmax[r] = fmaxf(tmax[r], __shfl_xor(tmax[r], msk, 32));
        }

        // ---- online softmax rescale ----
        float alpha[8];
#pragma unroll
        for (int r = 0; r < 8; ++r) {
            float mnew = fmaxf(mrun[r], tmax[r]);
            alpha[r] = __expf(mrun[r] - mnew);
            mrun[r] = mnew;
            lrun[r] *= alpha[r];
        }
#pragma unroll
        for (int t = 0; t < 4; ++t)
#pragma unroll
            for (int r = 0; r < 8; ++r) o[t][r] *= alpha[r];

        // ---- P = exp(S - m); row sums; stage P (bf16) to own LDS region ----
        float rsum[8] = {};
#pragma unroll
        for (int j = 0; j < 8; ++j)
#pragma unroll
            for (int r = 0; r < 8; ++r) {
                float p = __expf(s[j][r] - mrun[r]);
                rsum[r] += p;
                pw[(r + 8 * lh) * PST + j * 16 + ln] = (bf16)p;
            }
#pragma unroll
        for (int r = 0; r < 8; ++r) {
#pragma unroll
            for (int msk = 1; msk <= 8; msk <<= 1)
                rsum[r] += __shfl_xor(rsum[r], msk, 32);
            lrun[r] += rsum[r];
        }
        asm volatile("s_wait_dscnt 0" ::: "memory");

        // ---- O += P V : A = P (16x128) from LDS, B = V^T slice from LDS ----
#pragma unroll
        for (int ks = 0; ks < 4; ++ks) {
            const bf16* pp = pw + ln * PST + ks * 32;
            v16bf pa = pack16(pp + 8 * lh, pp + 16 + 8 * lh);
#pragma unroll
            for (int t = 0; t < 4; ++t) {
                const bf16* vp = Vbuf + (t * 16 + ln) * VST + ks * 32 + 8 * lh;
                v16bf bv = pack16(vp, vp + 16);
                o[t] = wmma_bf16(pa, bv, o[t]);
            }
        }
        __syncthreads();   // all waves done with buf before it is re-filled
    }

    // ---- normalize and write [B, C, H*hd] fp32 ----
#pragma unroll
    for (int t = 0; t < 4; ++t)
#pragma unroll
        for (int r = 0; r < 8; ++r) {
            int q = qbase + r + 8 * lh;
            attnO[((size_t)b * C + q) * 1024 + h * 64 + t * 16 + ln] =
                o[t][r] / lrun[r];
        }
}

// ---------------------------------------------------------------------------
// Launch. Workspace layout (48 MB):
//   [0,6M) Wqkv bf16  [6M,8M) Wo bf16  [8M,16M) Q  [16M,24M) K  [24M,32M) V^T
//   [32M,48M) attention output fp32
// ---------------------------------------------------------------------------
extern "C" void kernel_launch(void* const* d_in, const int* in_sizes, int n_in,
                              void* d_out, int out_size, void* d_ws,
                              size_t ws_size, hipStream_t stream) {
    const float* hidden = (const float*)d_in[0];
    const float* amask  = (const float*)d_in[1];
    const float* Wqkv   = (const float*)d_in[2];
    const float* bqkv   = (const float*)d_in[3];
    const float* Wo     = (const float*)d_in[4];
    const float* bo     = (const float*)d_in[5];
    float* out = (float*)d_out;

    constexpr int Bb = 2, C = 2048, D = 1024;
    constexpr int M = Bb * C;

    char* ws = (char*)d_ws;
    bf16* WqkvB = (bf16*)(ws);
    bf16* WoB   = (bf16*)(ws + (size_t)6 * 1024 * 1024);
    bf16* Qb    = (bf16*)(ws + (size_t)8 * 1024 * 1024);
    bf16* Kb    = (bf16*)(ws + (size_t)16 * 1024 * 1024);
    bf16* VTb   = (bf16*)(ws + (size_t)24 * 1024 * 1024);
    float* attn = (float*)(ws + (size_t)32 * 1024 * 1024);

    cvt_f32_to_bf16<<<dim3((D * 3 * D / 4 + 255) / 256), 256, 0, stream>>>(
        Wqkv, WqkvB, D * 3 * D);
    cvt_f32_to_bf16<<<dim3((D * D / 4 + 255) / 256), 256, 0, stream>>>(
        Wo, WoB, D * D);

    gemm_bf16<1><<<dim3(3 * D / BN, M / BM), 256, 0, stream>>>(
        hidden, WqkvB, bqkv, nullptr, Qb, Kb, VTb, M, 3 * D, D);

    attn_fwd<<<dim3(C / 128, Bb * 16), 256, 0, stream>>>(Qb, Kb, VTb, amask,
                                                         attn);

    gemm_bf16<0><<<dim3(D / BN, M / BM), 256, 0, stream>>>(
        attn, WoB, bo, out, nullptr, nullptr, nullptr, M, D, D);
}
